// MultiHeadSelfAttention_44865228374269
// MI455X (gfx1250) — matrix operand order repaired and checked by reading, not compiled, not run on hardware
//
#include <hip/hip_runtime.h>

// ---------------------------------------------------------------------------
// MHA forward for MI455X (gfx1250): bf16 WMMA (v_wmma_f32_16x16x32_bf16),
// f32 accumulate, flash-attention with async (ASYNCcnt) global->LDS staging of
// K/V tiles, block-cooperative LDS weight panels for the GEMMs, wave32.
// ---------------------------------------------------------------------------

typedef __bf16 bf16;
typedef __attribute__((ext_vector_type(16))) __bf16 v16bf;
typedef __attribute__((ext_vector_type(8)))  __bf16 v8bf;
typedef __attribute__((ext_vector_type(4)))  __bf16 v4bf;
typedef __attribute__((ext_vector_type(8)))  float  v8f;
typedef __attribute__((ext_vector_type(4)))  float  v4f;

#define WMMA_BF16(a, b, c) \
  __builtin_amdgcn_wmma_f32_16x16x32_bf16(false, (a), false, (b), (short)0, (c), false, false)

namespace {
constexpr int kB  = 4;
constexpr int kS  = 4096;
constexpr int kD  = 256;
constexpr int kH  = 4;
constexpr int kDh = 64;
constexpr int kRows = kB * kS;      // 16384
}  // namespace

__device__ __forceinline__ bf16 f2b(float f) { return (bf16)f; }

__device__ __forceinline__ v8f vzero8() {
  v8f z;
#pragma unroll
  for (int i = 0; i < 8; ++i) z[i] = 0.0f;
  return z;
}

// Async DMA: copy 16 bytes global -> LDS, tracked with ASYNCcnt (no VGPR round
// trip). dsaddr = low 32 bits of the generic LDS pointer (LDS aperture offset).
__device__ __forceinline__ void async_cp16(const bf16* gsrc, bf16* ldst) {
  unsigned dsaddr = (unsigned)(uintptr_t)ldst;
  unsigned long long gaddr = (unsigned long long)(uintptr_t)gsrc;
  asm volatile("global_load_async_to_lds_b128 %0, %1, off"
               :: "v"(dsaddr), "v"(gaddr) : "memory");
}

__device__ __forceinline__ void wait_async0() {
  asm volatile("s_wait_asynccnt 0x0" ::: "memory");
}

// A operand (16x32 bf16, MxK): lane holds row m = lane&15.
// ISA layout: lanes 0-15 -> K {kbase+0..7, kbase+16..23}, lanes 16-31 -> +8.
__device__ __forceinline__ v16bf load_a16(const bf16* __restrict__ row0, int ld,
                                          int lane, int kbase) {
  const int m  = lane & 15;
  const int hi = (lane >> 4) & 1;
  const bf16* p = row0 + (size_t)m * ld + kbase + hi * 8;
  v8bf c0 = *(const v8bf*)(p);
  v8bf c1 = *(const v8bf*)(p + 16);
  v16bf a;
#pragma unroll
  for (int i = 0; i < 8; ++i) { a[i] = c0[i]; a[8 + i] = c1[i]; }
  return a;
}

// B operand (32x16 bf16, KxN) read from B^T storage (row n holds K contiguous):
// lane holds column n = lane&15; lanes 0-15 -> K {kbase+0..7, kbase+16..23}, lanes 16-31 -> +8.
__device__ __forceinline__ v16bf load_b16(const bf16* __restrict__ bt0, int ld,
                                          int lane, int kbase) {
  const int n  = lane & 15;
  const int hi = (lane >> 4) & 1;
  const bf16* p = bt0 + (size_t)n * ld + kbase + hi * 8;
  v8bf c0 = *(const v8bf*)(p);
  v8bf c1 = *(const v8bf*)(p + 16);
  v16bf b;
#pragma unroll
  for (int i = 0; i < 8; ++i) { b[i] = c0[i]; b[8 + i] = c1[i]; }
  return b;
}

// One wave: 16x64 output tile. A streamed from global with a one-slice register
// prefetch; B served from the block's LDS weight panel (64 rows x 256).
__device__ __forceinline__ void gemm16x64_lds(const bf16* __restrict__ arow,
                                              const bf16* lpanel, int lane,
                                              v8f acc[4]) {
  v16bf a_cur = load_a16(arow, kD, lane, 0);
#pragma unroll
  for (int kt = 0; kt < 8; ++kt) {
    v16bf a_nxt = a_cur;
    if (kt < 7) a_nxt = load_a16(arow, kD, lane, (kt + 1) * 32);
#pragma unroll
    for (int t = 0; t < 4; ++t) {
      v16bf b = load_b16(lpanel + (size_t)(t * 16) * kD, kD, lane, kt * 32);
      acc[t] = WMMA_BF16(a_cur, b, acc[t]);
    }
    a_cur = a_nxt;
  }
}

// Stage a 64x256 W^T panel (32KB, contiguous in global) into LDS with async
// DMA: 256 threads x 8 slices x 16B, coalesced.
__device__ __forceinline__ void stage_wpanel(const bf16* wsrc, bf16* lpanel,
                                             int tid) {
#pragma unroll
  for (int r = 0; r < 8; ++r) {
    const int e = (r * 256 + tid) * 8;   // element offset, 16B granules
    async_cp16(wsrc + e, lpanel + e);
  }
  wait_async0();
  __syncthreads();
}

// ------------------------------ prep kernels -------------------------------

__global__ __launch_bounds__(256) void mha_cvt_x(const float* __restrict__ x,
                                                 bf16* __restrict__ xb) {
  const size_t i = ((size_t)blockIdx.x * blockDim.x + threadIdx.x) * 4;
  v4f v = *(const v4f*)(x + i);
  v4bf o;
#pragma unroll
  for (int j = 0; j < 4; ++j) o[j] = f2b(v[j]);
  *(v4bf*)(xb + i) = o;
}

// Wt[n][k] = bf16(W[k][n]) for the 4 weight matrices (gridDim.y selects).
__global__ __launch_bounds__(256) void mha_cvt_w(const float* __restrict__ wq,
                                                 const float* __restrict__ wk,
                                                 const float* __restrict__ wv,
                                                 const float* __restrict__ wo,
                                                 bf16* __restrict__ wqt,
                                                 bf16* __restrict__ wkt,
                                                 bf16* __restrict__ wvt,
                                                 bf16* __restrict__ wot) {
  const int n = blockIdx.x;     // 0..255 output row of W^T
  const int k = threadIdx.x;    // 0..255
  const float* src;
  bf16* dst;
  switch (blockIdx.y) {
    case 0:  src = wq; dst = wqt; break;
    case 1:  src = wk; dst = wkt; break;
    case 2:  src = wv; dst = wvt; break;
    default: src = wo; dst = wot; break;
  }
  dst[(size_t)n * kD + k] = f2b(src[(size_t)k * kD + n]);
}

// ------------------------------ QKV projection -----------------------------
// 1536 blocks: blk -> {mat 0..2, head nt 0..3, m-group 0..127}. The block's 8
// waves share one 64-col W^T panel staged in LDS; wave w does m-tile mg*8+w.
// Q,K stored row-major [b,h,s,Dh]; V stored transposed [b,h,Dh,s].
__global__ __launch_bounds__(256) void mha_qkv(
    const bf16* __restrict__ xb, const bf16* __restrict__ wqt,
    const bf16* __restrict__ wkt, const bf16* __restrict__ wvt,
    const float* __restrict__ bq, const float* __restrict__ bk,
    const float* __restrict__ bv, bf16* __restrict__ q, bf16* __restrict__ k,
    bf16* __restrict__ vt) {
  __shared__ __align__(16) bf16 lW[64 * kD];   // 32KB weight panel

  const int tid  = threadIdx.x;
  const int warp = tid >> 5;
  const int lane = tid & 31;
  const int blk  = blockIdx.x;
  const int mat  = blk >> 9;          // 512 blocks per matrix
  const int rem  = blk & 511;
  const int nt   = rem >> 7;          // head index (64-col panel)
  const int mt   = (rem & 127) * 8 + warp;

  const bf16* bt;
  const float* bias;
  if (mat == 0)      { bt = wqt; bias = bq; }
  else if (mat == 1) { bt = wkt; bias = bk; }
  else               { bt = wvt; bias = bv; }

  stage_wpanel(bt + (size_t)nt * 64 * kD, lW, tid);

  v8f acc[4] = {vzero8(), vzero8(), vzero8(), vzero8()};
  gemm16x64_lds(xb + (size_t)mt * 16 * kD, lW, lane, acc);

  const int n  = lane & 15;
  const int hi = (lane >> 4) & 1;
  const int b  = mt >> 8;              // 256 m-tiles per batch
  const int s0 = (mt & 255) * 16;
  const size_t bh = (size_t)(b * kH + nt);

#pragma unroll
  for (int t = 0; t < 4; ++t) {
    const float bsv = bias[nt * 64 + t * 16 + n];
    if (mat < 2) {
      bf16* dst = (mat == 0 ? q : k) +
                  (bh * kS + s0 + 8 * hi) * kDh + t * 16 + n;
#pragma unroll
      for (int v = 0; v < 8; ++v) dst[(size_t)v * kDh] = f2b(acc[t][v] + bsv);
    } else {
      // V^T[b,h, d = t*16+n, s = s0+8*hi+v]: contiguous in v -> 16B store.
      bf16* dst = vt + (bh * kDh + t * 16 + n) * kS + s0 + 8 * hi;
      v8bf pk;
#pragma unroll
      for (int v = 0; v < 8; ++v) pk[v] = f2b(acc[t][v] + bsv);
      *(v8bf*)dst = pk;
    }
  }
}

// ------------------------------ attention core -----------------------------
// 512 blocks; block = 8 waves = 128 queries of one (b,h). K/V tiles for each
// 32-key block are DMA'd into double-buffered LDS with async global->LDS ops:
// the copies for tile j+32 are issued first, the whole tile-j compute runs
// from LDS (hiding HBM/L2 latency behind WMMA+softmax), then each wave waits
// its own ASYNCcnt and the block barrier flips buffers. K/V global traffic is
// 8x lower than per-wave streaming (4 GB -> 512 MB).
__global__ __launch_bounds__(256) void mha_attn(const bf16* __restrict__ q,
                                                const bf16* __restrict__ kk,
                                                const bf16* __restrict__ vt,
                                                bf16* __restrict__ attn) {
  __shared__ __align__(16) bf16 lK[2][32 * 64];   // [key][dh]      4KB x2
  __shared__ __align__(16) bf16 lV[2][64 * 32];   // [dh][key]      4KB x2
  __shared__ __align__(16) bf16 lP[8][16 * 32];   // per-wave P     8KB

  const int tid  = threadIdx.x;
  const int warp = tid >> 5;
  const int lane = tid & 31;
  const int bh   = (int)blockIdx.x >> 5;          // 32 blocks per (b,h)
  const int b    = bh >> 2;
  const int h    = bh & 3;
  const int q0   = ((int)blockIdx.x & 31) * 128 + warp * 16;

  const bf16* qp = q  + ((size_t)bh * kS + q0) * kDh;
  const bf16* kp = kk + (size_t)bh * kS * kDh;
  const bf16* vp = vt + (size_t)bh * kDh * kS;

  const v16bf aq0 = load_a16(qp, kDh, lane, 0);
  const v16bf aq1 = load_a16(qp, kDh, lane, 32);

  v8f o[4] = {vzero8(), vzero8(), vzero8(), vzero8()};
  float mrun[8], lrun[8];
#pragma unroll
  for (int v = 0; v < 8; ++v) { mrun[v] = -1.0e30f; lrun[v] = 0.0f; }

  bf16* pl = lP[warp];
  const int n  = lane & 15;
  const int hi = (lane >> 4) & 1;
  const float scale = 0.125f;  // 1/sqrt(Dh)

  // This thread's 16B staging slices: K tile is 4KB contiguous (rows of 64
  // contiguous dh values); V tile is 64 rows x 32 keys from V^T (4 threads/row).
  const int vr = tid >> 2;
  const int vc = (tid & 3) * 8;

  // Prime buffer 0 with the j=0 tiles via async DMA.
  async_cp16(kp + (size_t)tid * 8, lK[0] + tid * 8);
  async_cp16(vp + (size_t)vr * kS + vc, lV[0] + vr * 32 + vc);
  wait_async0();
  __syncthreads();

  int buf = 0;
  for (int j = 0; j < kS; j += 32) {
    const bool pre = (j + 32 < kS);
    if (pre) {  // kick off next tile's DMA; it fills buf^1 during compute
      async_cp16(kp + (size_t)(j + 32) * kDh + (size_t)tid * 8,
                 lK[buf ^ 1] + tid * 8);
      async_cp16(vp + (size_t)vr * kS + (j + 32) + vc,
                 lV[buf ^ 1] + vr * 32 + vc);
    }
    const bf16* Kb = lK[buf];
    const bf16* Vb = lV[buf];

    // ---- scores: two 16x16 tiles covering keys [j, j+32), all from LDS ----
    v8f s0 = vzero8(), s1 = vzero8();
    s0 = WMMA_BF16(aq0, load_b16(Kb, kDh, lane, 0), s0);
    s0 = WMMA_BF16(aq1, load_b16(Kb, kDh, lane, 32), s0);
    s1 = WMMA_BF16(aq0, load_b16(Kb + 16 * kDh, kDh, lane, 0), s1);
    s1 = WMMA_BF16(aq1, load_b16(Kb + 16 * kDh, kDh, lane, 32), s1);

    // ---- online softmax over the 32-key block ------------------------------
#pragma unroll
    for (int v = 0; v < 8; ++v) {
      const float a0 = s0[v] * scale;
      const float a1 = s1[v] * scale;
      float t = fmaxf(a0, a1);
#pragma unroll
      for (int msk = 1; msk < 16; msk <<= 1) t = fmaxf(t, __shfl_xor(t, msk, 32));
      const float nm   = fmaxf(mrun[v], t);
      const float corr = __expf(mrun[v] - nm);
      mrun[v] = nm;
      const float p0 = __expf(a0 - nm);
      const float p1 = __expf(a1 - nm);
      float rs = p0 + p1;
#pragma unroll
      for (int msk = 1; msk < 16; msk <<= 1) rs += __shfl_xor(rs, msk, 32);
      lrun[v] = lrun[v] * corr + rs;
#pragma unroll
      for (int t4 = 0; t4 < 4; ++t4) o[t4][v] *= corr;
      const int m = v + 8 * hi;
      pl[m * 32 + n]      = f2b(p0);   // keys j+n
      pl[m * 32 + 16 + n] = f2b(p1);   // keys j+16+n
    }

    // Per-wave LDS store->load: DS pipe is in-order per wave; wait for stores.
    asm volatile("s_wait_dscnt 0" ::: "memory");
    const v16bf ap = load_a16(pl, 32, lane, 0);

    // ---- O += P @ V (V tile rows are contiguous keys in LDS) ---------------
#pragma unroll
    for (int t4 = 0; t4 < 4; ++t4) {
      o[t4] = WMMA_BF16(ap, load_b16(Vb + (size_t)(t4 * 16) * 32, 32, lane, 0),
                        o[t4]);
    }

    // Each wave waits its own async DMA, then the block barrier flips buffers.
    if (pre) wait_async0();
    __syncthreads();
    buf ^= 1;
  }

  // ---- normalize and write combined-heads attn[b, s, h*64 + d] (bf16) -----
  float inv[8];
#pragma unroll
  for (int v = 0; v < 8; ++v) inv[v] = 1.0f / lrun[v];

#pragma unroll
  for (int t4 = 0; t4 < 4; ++t4) {
    bf16* dst = attn + ((size_t)b * kS + q0 + 8 * hi) * kD +
                h * 64 + t4 * 16 + n;
#pragma unroll
    for (int v = 0; v < 8; ++v) dst[(size_t)v * kD] = f2b(o[t4][v] * inv[v]);
  }
}

// ------------------------------ output projection --------------------------
// 512 blocks: blk -> {nt 0..3, m-group 0..127}; LDS W^T panel as in mha_qkv.
__global__ __launch_bounds__(256) void mha_oproj(const bf16* __restrict__ ab,
                                                 const bf16* __restrict__ wot,
                                                 const float* __restrict__ bo,
                                                 float* __restrict__ out) {
  __shared__ __align__(16) bf16 lW[64 * kD];   // 32KB weight panel

  const int tid  = threadIdx.x;
  const int warp = tid >> 5;
  const int lane = tid & 31;
  const int blk  = blockIdx.x;
  const int nt   = blk >> 7;
  const int mt   = (blk & 127) * 8 + warp;

  stage_wpanel(wot + (size_t)nt * 64 * kD, lW, tid);

  v8f acc[4] = {vzero8(), vzero8(), vzero8(), vzero8()};
  gemm16x64_lds(ab + (size_t)mt * 16 * kD, lW, lane, acc);

  const int n  = lane & 15;
  const int hi = (lane >> 4) & 1;
  const size_t r0 = (size_t)mt * 16 + 8 * hi;
#pragma unroll
  for (int t = 0; t < 4; ++t) {
    const float bsv = bo[nt * 64 + t * 16 + n];
    float* dst = out + r0 * kD + nt * 64 + t * 16 + n;
#pragma unroll
    for (int v = 0; v < 8; ++v) dst[(size_t)v * kD] = acc[t][v] + bsv;
  }
}

// ------------------------------ host launcher ------------------------------

extern "C" void kernel_launch(void* const* d_in, const int* in_sizes, int n_in,
                              void* d_out, int out_size, void* d_ws, size_t ws_size,
                              hipStream_t stream) {
  const float* x  = (const float*)d_in[0];
  const float* wq = (const float*)d_in[1];
  const float* bq = (const float*)d_in[2];
  const float* wk = (const float*)d_in[3];
  const float* bk = (const float*)d_in[4];
  const float* wv = (const float*)d_in[5];
  const float* bv = (const float*)d_in[6];
  const float* wo = (const float*)d_in[7];
  const float* bo = (const float*)d_in[8];

  // Workspace layout (bytes), total ~40.5 MB.
  constexpr size_t kBigBytes = (size_t)kRows * kD * sizeof(bf16);  // 8 MB each
  constexpr size_t kWtBytes  = (size_t)kD * kD * sizeof(bf16);     // 128 KB each
  char* ws = (char*)d_ws;
  bf16* xb   = (bf16*)(ws);
  bf16* qb   = (bf16*)(ws + 1 * kBigBytes);
  bf16* kb   = (bf16*)(ws + 2 * kBigBytes);
  bf16* vtb  = (bf16*)(ws + 3 * kBigBytes);
  bf16* atb  = (bf16*)(ws + 4 * kBigBytes);
  bf16* wqt  = (bf16*)(ws + 5 * kBigBytes);
  bf16* wkt  = (bf16*)(ws + 5 * kBigBytes + 1 * kWtBytes);
  bf16* wvt  = (bf16*)(ws + 5 * kBigBytes + 2 * kWtBytes);
  bf16* wot  = (bf16*)(ws + 5 * kBigBytes + 3 * kWtBytes);

  // 1) convert inputs / transpose weights to bf16
  mha_cvt_x<<<(kRows * kD) / (256 * 4), 256, 0, stream>>>(x, xb);
  mha_cvt_w<<<dim3(kD, 4), 256, 0, stream>>>(wq, wk, wv, wo, wqt, wkt, wvt, wot);

  // 2) QKV projection: 3 mats x 4 heads x 128 m-groups = 1536 blocks
  mha_qkv<<<1536, 256, 0, stream>>>(xb, wqt, wkt, wvt, bq, bk, bv, qb, kb, vtb);

  // 3) flash attention: 16 (b,h) * 32 blocks (128 queries each) = 512 blocks
  mha_attn<<<512, 256, 0, stream>>>(qb, kb, vtb, atb);

  // 4) output projection: 4 nt x 128 m-groups = 512 blocks
  mha_oproj<<<512, 256, 0, stream>>>(atb, wot, bo, (float*)d_out);
}
